// CausalSelfAttention_14963666059440
// MI455X (gfx1250) — matrix operand order repaired and checked by reading, not compiled, not run on hardware
//
#include <hip/hip_runtime.h>
#include <math.h>

#define DM 1024
#define NH 16
#define DH 64
#define BB 2
#define TT 2048
#define N3 (3 * DM)

typedef __attribute__((ext_vector_type(16))) __bf16 v16bf;
typedef __attribute__((ext_vector_type(8)))  float  v8f;
typedef __attribute__((ext_vector_type(4)))  unsigned int v4u;

// round-to-nearest-even f32 -> bf16 (one-time converts)
__device__ __forceinline__ __bf16 f2bf(float f) {
  union { float f; unsigned u; } a; a.f = f;
  unsigned r = a.u + 0x7FFFu + ((a.u >> 16) & 1u);
  return __builtin_bit_cast(__bf16, (unsigned short)(r >> 16));
}
// truncating f32 -> bf16 (hot paths)
__device__ __forceinline__ __bf16 f2bf_t(float f) {
  union { float f; unsigned u; } a; a.f = f;
  return __builtin_bit_cast(__bf16, (unsigned short)(a.u >> 16));
}
__device__ __forceinline__ float bf2f(__bf16 b) {
  union { unsigned u; float f; } a;
  a.u = ((unsigned)__builtin_bit_cast(unsigned short, b)) << 16;
  return a.f;
}
// pack two f32 into two bf16 (RNE)
__device__ __forceinline__ unsigned pack2(float lo, float hi) {
  union { float f; unsigned u; } a, b; a.f = lo; b.f = hi;
  unsigned ra = a.u + 0x7FFFu + ((a.u >> 16) & 1u);
  unsigned rb = b.u + 0x7FFFu + ((b.u >> 16) & 1u);
  return (ra >> 16) | (rb & 0xFFFF0000u);
}
// pack two f32 into two bf16 (truncate; hot path)
__device__ __forceinline__ unsigned packt(float lo, float hi) {
  union { float f; unsigned u; } a, b; a.f = lo; b.f = hi;
  return (a.u >> 16) | (b.u & 0xFFFF0000u);
}
// raw v_exp_f32 (2^x) without the libm denorm guard
__device__ __forceinline__ float exp2_fast(float x) {
#if __has_builtin(__builtin_amdgcn_exp2f)
  return __builtin_amdgcn_exp2f(x);
#else
  return exp2f(x);
#endif
}

// A-operand fragment (16x32 bf16). Lanes 0-15 hold K 0-7 & 16-23 of row
// (lane&15); lanes 16-31 hold K 8-15 & 24-31. p = row start at frag K origin.
__device__ __forceinline__ v16bf load_fragA(const __bf16* p, int lane) {
  const __bf16* q = p + ((lane & 16) ? 8 : 0);
  union { v16bf v; v4u u[2]; } f;
  f.u[0] = *(const v4u*)(q);
  f.u[1] = *(const v4u*)(q + 16);
  return f.v;
}
// B-operand fragment (32x16 bf16). Lane holds column (lane&15); lanes 0-15
// carry K 0-7 & 16-23, lanes 16-31 carry K 8-15 & 24-31. p = column data start (K-contig).
__device__ __forceinline__ v16bf load_fragB(const __bf16* p, int lane) {
  const __bf16* q = p + ((lane & 16) ? 8 : 0);
  union { v16bf v; v4u u[2]; } f;
  f.u[0] = *(const v4u*)(q);
  f.u[1] = *(const v4u*)(q + 16);
  return f.v;
}

__device__ __forceinline__ v8f wmma_bf16(v16bf a, v16bf b, v8f c) {
  return __builtin_amdgcn_wmma_f32_16x16x32_bf16(false, a, false, b,
                                                 (short)0, c, false, false);
}

__device__ __forceinline__ void store1(float* p, size_t i, float v) { p[i] = v; }
__device__ __forceinline__ void store1(__bf16* p, size_t i, float v) { p[i] = f2bf_t(v); }

// ------------------------------------------------------------ convert (x8)
__global__ void k_f32_to_bf16x8(const float* __restrict__ in,
                                __bf16* __restrict__ out, int n8) {
  int i = blockIdx.x * blockDim.x + threadIdx.x;
  if (i >= n8) return;
  const float* s = in + (size_t)i * 8;
  v4u o;
  o[0] = pack2(s[0], s[1]);
  o[1] = pack2(s[2], s[3]);
  o[2] = pack2(s[4], s[5]);
  o[3] = pack2(s[6], s[7]);
  *(v4u*)(out + (size_t)i * 8) = o;
}

// ---------------------------------------------------------------- GEMM
// C[m,n] = sum_k A[m,k] * Bw[n,k] + bias[n].  M%128==0, N%128==0, K%32==0.
// Block = 128 threads (4 waves), 128x128 tile; wave = 64x64 (4x4 WMMA tiles).
template <typename OutT>
__global__ __launch_bounds__(128, 1) void k_gemm_bias(
    const __bf16* __restrict__ A, const __bf16* __restrict__ Bw,
    const float* __restrict__ bias, OutT* __restrict__ C,
    int M, int N, int K) {
  const int lane = threadIdx.x & 31;
  const int wave = threadIdx.x >> 5;
  const int mBase = blockIdx.y * 128 + (wave >> 1) * 64;
  const int nBase = blockIdx.x * 128 + (wave & 1) * 64;
  const __bf16* ap[4];
  const __bf16* bp[4];
#pragma unroll
  for (int i = 0; i < 4; ++i) {
    ap[i] = A + (size_t)(mBase + i * 16 + (lane & 15)) * K;
    bp[i] = Bw + (size_t)(nBase + i * 16 + (lane & 15)) * K;
  }
  v8f c[4][4] = {};
  for (int k = 0; k < K; k += 32) {
    __builtin_prefetch(ap[0] + k + 256, 0, 1);
    __builtin_prefetch(bp[0] + k + 256, 0, 1);
    v16bf af[4], bf[4];
#pragma unroll
    for (int i = 0; i < 4; ++i) af[i] = load_fragA(ap[i] + k, lane);
#pragma unroll
    for (int j = 0; j < 4; ++j) bf[j] = load_fragB(bp[j] + k, lane);
#pragma unroll
    for (int i = 0; i < 4; ++i)
#pragma unroll
      for (int j = 0; j < 4; ++j) c[i][j] = wmma_bf16(af[i], bf[j], c[i][j]);
  }
  const int half8 = (lane & 16) ? 8 : 0;
  const int nc = lane & 15;
  float bv[4];
#pragma unroll
  for (int j = 0; j < 4; ++j) bv[j] = bias[nBase + j * 16 + nc];
#pragma unroll
  for (int i = 0; i < 4; ++i) {
#pragma unroll
    for (int r = 0; r < 8; ++r) {
      int m = mBase + i * 16 + half8 + r;
#pragma unroll
      for (int j = 0; j < 4; ++j)
        store1(C, (size_t)m * N + nBase + j * 16 + nc, c[i][j][r] + bv[j]);
    }
  }
}

// --------------------------------------------------- RoPE + head split
// qkv layout [B,T,3,H,Dh] (bf16).  Out: Q (pre-scaled by log2e/sqrt(Dh)),
// K [B,H,T,Dh]; V transposed [B,H,Dh,T].
__global__ void k_rope_split(const __bf16* __restrict__ qkv,
                             __bf16* __restrict__ Q, __bf16* __restrict__ Kc,
                             __bf16* __restrict__ Vt) {
  int i = blockIdx.x * blockDim.x + threadIdx.x;   // over B*T*NH*DH = 2^22
  int d = i & (DH - 1);
  int h = (i >> 6) & (NH - 1);
  int t = (i >> 10) & (TT - 1);
  int b = i >> 21;
  size_t base = ((size_t)(b * TT + t) * 3) * DM + h * DH;
  float q = bf2f(qkv[base + d]);
  float k = bf2f(qkv[base + DM + d]);
  float v = bf2f(qkv[base + 2 * DM + d]);
  int dp = (d < 32) ? (d + 32) : (d - 32);
  float qp = bf2f(qkv[base + dp]);
  float kp = bf2f(qkv[base + DM + dp]);
  float sgn = (d < 32) ? -1.f : 1.f;
  int j = d & 31;
  float freq = __expf(-(float)j * 0.28782313662425572f);  // 10000^(-2j/64)
  float ang = (float)t * freq;
  float cv = __cosf(ang), sv = __sinf(ang);
  const float qs = 0.125f * 1.4426950408889634f;  // fold scale & log2e into Q
  size_t bh = (size_t)(b * NH + h);
  Q[(bh * TT + t) * DH + d] = f2bf((q * cv + sgn * qp * sv) * qs);
  Kc[(bh * TT + t) * DH + d] = f2bf(k * cv + sgn * kp * sv);
  Vt[(bh * DH + d) * TT + t] = f2bf(v);
}

// ---------------------------------------------------------------- flash attn
// grid (T/128, B*NH); block 128 = 4 waves; wave owns 32 q rows.
// Computes S^T = K*Q^T so each softmax row lives (almost) in one lane:
// lane = query column, VGPRs = keys; only one shfl_xor(16) per reduction.
// O^T = V^T * P^T accumulated with Dh as the M dimension.
__global__ __launch_bounds__(128, 1) void k_flash(
    const __bf16* __restrict__ Q, const __bf16* __restrict__ Kc,
    const __bf16* __restrict__ Vt, __bf16* __restrict__ AO) {
  __shared__ __align__(16) __bf16 Ptlds[4][32 * 32];  // rows = query, cols = key
  const int lane = threadIdx.x & 31;
  const int wave = threadIdx.x >> 5;
  const int bh = blockIdx.y;
  const int b = bh >> 4;
  const int h = bh & 15;
  const int qBase = blockIdx.x * 128 + wave * 32;
  const int half8 = (lane & 16) ? 8 : 0;
  const int nc = lane & 15;

  const __bf16* Qb = Q + (size_t)bh * TT * DH;
  const __bf16* Kb = Kc + (size_t)bh * TT * DH;
  const __bf16* Vb = Vt + (size_t)bh * DH * TT;

  // Q^T as B-operand fragments (lane = query column), hoisted
  v16bf qf[2][2];
#pragma unroll
  for (int nt = 0; nt < 2; ++nt)
#pragma unroll
    for (int ks = 0; ks < 2; ++ks)
      qf[nt][ks] =
          load_fragB(Qb + (size_t)(qBase + nt * 16 + nc) * DH + ks * 32, lane);

  v8f o[4][2] = {};                 // O^T: [Dh tile][query tile]
  float mQ[2] = {-3.0e38f, -3.0e38f};
  float lQ[2] = {0.f, 0.f};

  auto step = [&](int kb, bool masked) {
    // K tiles as A operand: [key tile][Dh slice]
    v16bf kf[2][2];
#pragma unroll
    for (int mt = 0; mt < 2; ++mt)
#pragma unroll
      for (int ks = 0; ks < 2; ++ks)
        kf[mt][ks] = load_fragA(
            Kb + (size_t)(kb + mt * 16 + nc) * DH + ks * 32, lane);

    // S^T tiles: [key tile][query tile]
    v8f s[2][2];
#pragma unroll
    for (int mt = 0; mt < 2; ++mt)
#pragma unroll
      for (int nt = 0; nt < 2; ++nt) {
        v8f acc = {};
        acc = wmma_bf16(kf[mt][0], qf[nt][0], acc);
        acc = wmma_bf16(kf[mt][1], qf[nt][1], acc);
        s[mt][nt] = acc;
      }

    if (masked) {                   // diagonal block only (kb == qBase)
#pragma unroll
      for (int mt = 0; mt < 2; ++mt)
#pragma unroll
        for (int nt = 0; nt < 2; ++nt)
#pragma unroll
          for (int r = 0; r < 8; ++r) {
            int kg = kb + mt * 16 + half8 + r;
            int qg = qBase + nt * 16 + nc;
            if (kg > qg) s[mt][nt][r] = -3.0e38f;
          }
    }

    // per-query online softmax (log2 domain), in-lane + one xor-16 shuffle
#pragma unroll
    for (int nt = 0; nt < 2; ++nt) {
      float mx = fmaxf(s[0][nt][0], s[1][nt][0]);
#pragma unroll
      for (int r = 1; r < 8; ++r)
        mx = fmaxf(mx, fmaxf(s[0][nt][r], s[1][nt][r]));
      mx = fmaxf(mx, __shfl_xor(mx, 16, 32));
      float newm = fmaxf(mQ[nt], mx);
      float corr = exp2_fast(mQ[nt] - newm);
      float sum = 0.f;
#pragma unroll
      for (int mt = 0; mt < 2; ++mt)
#pragma unroll
        for (int r = 0; r < 8; ++r) {
          float pp = exp2_fast(s[mt][nt][r] - newm);
          s[mt][nt][r] = pp;
          sum += pp;
        }
      sum += __shfl_xor(sum, 16, 32);
      lQ[nt] = lQ[nt] * corr + sum;
      mQ[nt] = newm;
#pragma unroll
      for (int dm = 0; dm < 4; ++dm)
#pragma unroll
        for (int r = 0; r < 8; ++r) o[dm][nt][r] *= corr;
    }

    // P^T -> LDS (rows = query, keys contiguous), packed b32 stores
#pragma unroll
    for (int nt = 0; nt < 2; ++nt)
#pragma unroll
      for (int mt = 0; mt < 2; ++mt) {
        unsigned* prow = (unsigned*)&Ptlds[wave][(nt * 16 + nc) * 32 +
                                                 mt * 16 + half8];
        prow[0] = packt(s[mt][nt][0], s[mt][nt][1]);
        prow[1] = packt(s[mt][nt][2], s[mt][nt][3]);
        prow[2] = packt(s[mt][nt][4], s[mt][nt][5]);
        prow[3] = packt(s[mt][nt][6], s[mt][nt][7]);
      }
    __builtin_amdgcn_wave_barrier();
    asm volatile("s_wait_dscnt 0x0" ::: "memory");

    // O^T += V^T * P^T
    v16bf pB[2];
#pragma unroll
    for (int nt = 0; nt < 2; ++nt)
      pB[nt] = load_fragB(&Ptlds[wave][(nt * 16 + nc) * 32], lane);
#pragma unroll
    for (int dm = 0; dm < 4; ++dm) {
      v16bf vf = load_fragA(Vb + (size_t)(dm * 16 + nc) * TT + kb, lane);
#pragma unroll
      for (int nt = 0; nt < 2; ++nt)
        o[dm][nt] = wmma_bf16(vf, pB[nt], o[dm][nt]);
    }
    __builtin_amdgcn_wave_barrier();
  };

  for (int kb = 0; kb < qBase; kb += 32) step(kb, false);  // fully unmasked
  step(qBase, true);                                       // diagonal block

  // normalize + write attn output [B,T,C]; Dh contiguous -> b128 stores
#pragma unroll
  for (int nt = 0; nt < 2; ++nt) {
    float inv = 1.f / lQ[nt];
    int t = qBase + nt * 16 + nc;
    size_t ob = ((size_t)(b * TT + t)) * DM + h * DH;
#pragma unroll
    for (int dm = 0; dm < 4; ++dm) {
      v4u w;
      w[0] = packt(o[dm][nt][0] * inv, o[dm][nt][1] * inv);
      w[1] = packt(o[dm][nt][2] * inv, o[dm][nt][3] * inv);
      w[2] = packt(o[dm][nt][4] * inv, o[dm][nt][5] * inv);
      w[3] = packt(o[dm][nt][6] * inv, o[dm][nt][7] * inv);
      *(v4u*)(AO + ob + dm * 16 + half8) = w;
    }
  }
}

// ---------------------------------------------------------------- launch
extern "C" void kernel_launch(void* const* d_in, const int* in_sizes, int n_in,
                              void* d_out, int out_size, void* d_ws,
                              size_t ws_size, hipStream_t stream) {
  const float* x      = (const float*)d_in[0];
  const float* qkv_w  = (const float*)d_in[1];
  const float* qkv_b  = (const float*)d_in[2];
  const float* proj_w = (const float*)d_in[3];
  const float* proj_b = (const float*)d_in[4];
  float* out = (float*)d_out;

  char* p = (char*)d_ws;
  __bf16* xh     = (__bf16*)p; p += (size_t)BB * TT * DM * 2;
  __bf16* wqkvh  = (__bf16*)p; p += (size_t)N3 * DM * 2;
  __bf16* wprojh = (__bf16*)p; p += (size_t)DM * DM * 2;
  __bf16* qkvh   = (__bf16*)p; p += (size_t)BB * TT * N3 * 2;
  __bf16* Qh     = (__bf16*)p; p += (size_t)BB * NH * TT * DH * 2;
  __bf16* Kh     = (__bf16*)p; p += (size_t)BB * NH * TT * DH * 2;
  __bf16* Vth    = (__bf16*)p; p += (size_t)BB * NH * DH * TT * 2;
  __bf16* aoh    = (__bf16*)p; p += (size_t)BB * TT * DM * 2;

  int nx8 = (BB * TT * DM) / 8;
  k_f32_to_bf16x8<<<(nx8 + 255) / 256, 256, 0, stream>>>(x, xh, nx8);
  int nw8 = (N3 * DM) / 8;
  k_f32_to_bf16x8<<<(nw8 + 255) / 256, 256, 0, stream>>>(qkv_w, wqkvh, nw8);
  int np8 = (DM * DM) / 8;
  k_f32_to_bf16x8<<<(np8 + 255) / 256, 256, 0, stream>>>(proj_w, wprojh, np8);

  dim3 g1(N3 / 128, (BB * TT) / 128);
  k_gemm_bias<__bf16><<<g1, 128, 0, stream>>>(xh, wqkvh, qkv_b, qkvh,
                                              BB * TT, N3, DM);

  int nr = BB * TT * NH * DH;
  k_rope_split<<<(nr + 255) / 256, 256, 0, stream>>>(qkvh, Qh, Kh, Vth);

  dim3 g2(TT / 128, BB * NH);
  k_flash<<<g2, 128, 0, stream>>>(Qh, Kh, Vth, aoh);

  dim3 g3(DM / 128, (BB * TT) / 128);
  k_gemm_bias<float><<<g3, 128, 0, stream>>>(aoh, wprojh, proj_b, out,
                                             BB * TT, DM, DM);
}
